// TEGConv_24575802868350
// MI455X (gfx1250) — matrix operand order repaired and checked by reading, not compiled, not run on hardware
//
#include <hip/hip_runtime.h>

// ---------------------------------------------------------------------------
// TEGConv on gfx1250: per-edge [E,192]x[192,128] GEMM on the bf16 matrix pipe
// using the bf16x3 split (hi/lo) with f32 accumulation:
//   A*B ~= Ahi*Bhi + Ahi*Blo + Alo*Bhi        (~1e-4 rel err, f32 accumulate)
// 3 x v_wmma_f32_16x16x32_bf16 per K=32 slab replaces 8 x f32 K=4 WMMAs
// -> ~2.67x matrix throughput on the compute-bound GEMM.
// ---------------------------------------------------------------------------

#define N_NODES   10000
#define D_NODE    128
#define D_EDGE    64
#define D_IN      192      // D_NODE + D_EDGE
#define D_OUT     128

#define M_BLOCK   64       // edges per block tile
#define KPAD      200      // padded bf16 K-stride (192 -> 200, mult of 8)

typedef __attribute__((ext_vector_type(16))) __bf16 v16bf;
typedef __attribute__((ext_vector_type(8)))  float  v8f;

union Frag {               // 32 bytes: one bf16 WMMA A/B operand
  uint4 u[2];
  v16bf bf;
};

#define WMMA_BF16(A, B, C)                                                   \
  __builtin_amdgcn_wmma_f32_16x16x32_bf16(false, (A), false, (B), (short)0,  \
                                          (C), false, false)

__device__ __forceinline__ unsigned int f32_to_bf16_rne(float f) {
  unsigned int u = __float_as_uint(f);
  return (u + 0x7FFFu + ((u >> 16) & 1u)) >> 16;   // bf16 bits in [15:0]
}

__device__ __forceinline__ void split_bf16(float f, unsigned int& hi,
                                           unsigned int& lo) {
  hi = f32_to_bf16_rne(f);
  float fh = __uint_as_float(hi << 16);
  lo = f32_to_bf16_rne(f - fh);
}

__global__ void zero_kernel(float* __restrict__ p, int n) {
  int i = blockIdx.x * blockDim.x + threadIdx.x;
  if (i < n) p[i] = 0.0f;
}

__global__ __launch_bounds__(256) void tegconv_gemm(
    const float* __restrict__ x,       // [N_NODES, D_NODE]
    const int*   __restrict__ eidx,    // [2, E]  (src row, then dst row)
    const float* __restrict__ ef,      // [E, D_EDGE]
    const float* __restrict__ W,       // [D_IN, D_OUT]
    const float* __restrict__ bias,    // [D_OUT]
    float*       __restrict__ sums,    // [N_NODES, D_OUT]
    float*       __restrict__ cnt,     // [N_NODES]
    int n_edges)
{
  extern __shared__ char smem[];
  // bf16 planes; W stored TRANSPOSED (Wt[N][K]) so B fragments are contiguous.
  unsigned short* sWhi = (unsigned short*)smem;                       // 128*KPAD
  unsigned short* sWlo = sWhi + D_OUT * KPAD;
  unsigned short* sAhi = sWlo + D_OUT * KPAD;                         // 64*KPAD
  unsigned short* sAlo = sAhi + M_BLOCK * KPAD;
  int* sDst = (int*)(sAlo + M_BLOCK * KPAD);                          // [64]

  const int tid  = threadIdx.x;
  const int lane = tid & 31;
  const int wv   = tid >> 5;          // 0..7 (8 waves)

  // ---- Stage W transposed + hi/lo split, once per block (L2-resident) ----
  for (int idx = tid; idx < D_IN * D_OUT; idx += 256) {
    int k = idx >> 7;                 // W row-major [K][N]
    int n = idx & 127;
    unsigned int hi, lo;
    split_bf16(W[idx], hi, lo);
    sWhi[n * KPAD + k] = (unsigned short)hi;
    sWlo[n * KPAD + k] = (unsigned short)lo;
  }

  // A-staging role: 4 threads per edge row
  const int row = tid >> 2;           // 0..63
  const int sub = tid & 3;

  // Wave tile: wave wv owns edges [m0, m0+16) x outputs [nh*64, nh*64+64)
  const int mt = wv & 3;
  const int nh = wv >> 2;             // 0 or 1
  const int m0 = mt * 16;
  const int r  = lane & 15;           // M (A) / N (B) within tile
  const int hh = lane >> 4;           // half-wave selects K chunk

  const int nblocks = n_edges / M_BLOCK;
  for (int eb = blockIdx.x; eb < nblocks; eb += gridDim.x) {
    __syncthreads();  // epilogue of previous iteration done before restaging

    // ---- Stage A tile rows [x[src] | ef[e]] as bf16 hi/lo planes ----
    const int e_row = eb * M_BLOCK + row;
    if (sub == 0) {
      int dv = eidx[n_edges + e_row];
      sDst[row] = dv;
      atomicAdd(&cnt[dv], 1.0f);
    }
    const int src = eidx[e_row];
    const float* xrow  = x  + (long)src   * D_NODE;
    const float* efrow = ef + (long)e_row * D_EDGE;
#pragma unroll
    for (int j = 0; j < 12; ++j) {
      int c4 = sub + j * 4;           // float4 index 0..47
      int c  = c4 * 4;
      float4 v = (c < D_NODE) ? ((const float4*)xrow)[c4]
                              : ((const float4*)efrow)[c4 - (D_NODE / 4)];
      unsigned int h0, l0, h1, l1, h2, l2, h3, l3;
      split_bf16(v.x, h0, l0);
      split_bf16(v.y, h1, l1);
      split_bf16(v.z, h2, l2);
      split_bf16(v.w, h3, l3);
      uint2 hp = make_uint2(h0 | (h1 << 16), h2 | (h3 << 16));
      uint2 lp = make_uint2(l0 | (l1 << 16), l2 | (l3 << 16));
      *(uint2*)&sAhi[row * KPAD + c] = hp;
      *(uint2*)&sAlo[row * KPAD + c] = lp;
    }
    __syncthreads();

    // ---- 16x64 output tile per wave: 6 K-slabs x 4 N-tiles x 3 WMMAs ----
    v8f acc[4] = {v8f{}, v8f{}, v8f{}, v8f{}};
    const unsigned short* aRowHi = sAhi + (m0 + r) * KPAD;
    const unsigned short* aRowLo = sAlo + (m0 + r) * KPAD;
#pragma unroll 2
    for (int ks = 0; ks < D_IN; ks += 32) {
      // A 16x32 bf16 fragment: lane half hh -> K chunks {ca..ca+7, ca+16..+23}
      const int ca = ks + hh * 8;
      Frag ahi, alo;
      ahi.u[0] = *(const uint4*)(aRowHi + ca);
      ahi.u[1] = *(const uint4*)(aRowHi + ca + 16);
      alo.u[0] = *(const uint4*)(aRowLo + ca);
      alo.u[1] = *(const uint4*)(aRowLo + ca + 16);
      // B 32x16 bf16 fragment: lane half hh -> K chunks {cb..cb+7, cb+16..+23}
      const int cb = ks + hh * 8;
#pragma unroll
      for (int i = 0; i < 4; ++i) {
        const int n = nh * 64 + i * 16 + r;
        const unsigned short* bRowHi = sWhi + n * KPAD;
        const unsigned short* bRowLo = sWlo + n * KPAD;
        Frag bhi, blo;
        bhi.u[0] = *(const uint4*)(bRowHi + cb);
        bhi.u[1] = *(const uint4*)(bRowHi + cb + 16);
        blo.u[0] = *(const uint4*)(bRowLo + cb);
        blo.u[1] = *(const uint4*)(bRowLo + cb + 16);
        acc[i] = WMMA_BF16(ahi.bf, bhi.bf, acc[i]);   // hi*hi
        acc[i] = WMMA_BF16(ahi.bf, blo.bf, acc[i]);   // hi*lo
        acc[i] = WMMA_BF16(alo.bf, bhi.bf, acc[i]);   // lo*hi
      }
    }

    // ---- Epilogue: bias + atomic scatter into per-destination sums ----
    const int mrow_base = m0 + hh * 8;      // C layout: VGPR j -> row j (+8)
    const int ncol_base = r + nh * 64;
#pragma unroll
    for (int i = 0; i < 4; ++i) {
      const int n  = ncol_base + i * 16;
      const float bv = bias[n];
#pragma unroll
      for (int j = 0; j < 8; ++j) {
        const int m   = mrow_base + j;
        const int dst = sDst[m];
        atomicAdd(&sums[(long)dst * D_OUT + n], acc[i][j] + bv);
      }
    }
  }
}

__global__ void finalize_kernel(const float* __restrict__ sums,
                                const float* __restrict__ cnt,
                                float* __restrict__ out, int total) {
  int i = blockIdx.x * blockDim.x + threadIdx.x;
  if (i < total) {
    float c = cnt[i >> 7];            // i / D_OUT
    out[i] = sums[i] / fmaxf(c, 1.0f);
  }
}

extern "C" void kernel_launch(void* const* d_in, const int* in_sizes, int n_in,
                              void* d_out, int out_size, void* d_ws, size_t ws_size,
                              hipStream_t stream) {
  const float* x    = (const float*)d_in[0];
  const int*   eidx = (const int*)  d_in[1];   // harness convention: int32
  const float* ef   = (const float*)d_in[2];
  const float* W    = (const float*)d_in[3];
  const float* bias = (const float*)d_in[4];
  float* out = (float*)d_out;

  float* sums = (float*)d_ws;                  // N_NODES * D_OUT
  float* cnt  = sums + (size_t)N_NODES * D_OUT;
  const int n_edges = in_sizes[1] / 2;

  const int n_acc = N_NODES * D_OUT + N_NODES;
  zero_kernel<<<(n_acc + 255) / 256, 256, 0, stream>>>(sums, n_acc);

  const size_t smem_bytes =
      (size_t)(D_OUT * KPAD * 2 + M_BLOCK * KPAD * 2) * sizeof(unsigned short) +
      M_BLOCK * sizeof(int);                   // ~150 KB -> 2 WG per 320KB WGP
  tegconv_gemm<<<1024, 256, smem_bytes, stream>>>(x, eidx, ef, W, bias,
                                                  sums, cnt, n_edges);

  const int n_out = N_NODES * D_OUT;
  finalize_kernel<<<(n_out + 255) / 256, 256, 0, stream>>>(sums, cnt, out, n_out);
}
